// pc_conv_30159260352601
// MI455X (gfx1250) — matrix operand order repaired and checked by reading, not compiled, not run on hardware
//
#include <hip/hip_runtime.h>

typedef _Float16 f16;
typedef __attribute__((ext_vector_type(16))) _Float16 v16h;
typedef __attribute__((ext_vector_type(8)))  _Float16 v8h;
typedef __attribute__((ext_vector_type(8)))  float    v8f;
typedef __attribute__((ext_vector_type(4)))  float    v4f;

#define EF   128
#define K1P  160              // layer-1 K padded: 128 feat + 3 xyz + 1 bias + 28 zeros
#define ROWS_PER_WAVE 32      // two 16-row A tiles share every B fragment
#define ROWS_PER_BLOCK 256    // 8 waves * 32 rows = 32 points per tile-iteration

union AFrag { v16h v; v8h h[2]; };

__global__ __launch_bounds__(256) void pc_conv_wmma_kernel(
    const float* __restrict__ input,     // [N, 128]
    const int*   __restrict__ knn_idx,   // [N*8]
    const float* __restrict__ knn_xyz,   // [N*8, 3]
    const float* __restrict__ W1,        // [131, 128] row-major
    const float* __restrict__ b1,        // [128]
    const float* __restrict__ W2,        // [128, 128] row-major
    const float* __restrict__ b2,        // [128]
    float* __restrict__ out,             // [N, 128]
    int numTiles)
{
    __shared__ __attribute__((aligned(32))) f16  sW1[K1P * EF];     // 40 KB, K-padded
    __shared__ __attribute__((aligned(32))) f16  sW2[EF * EF];      // 32 KB
    __shared__ __attribute__((aligned(32))) f16  sH[8][32 * EF];    // 64 KB (per-wave h1 tile)
    __shared__ float sB2[EF];                                       // layer-2 bias (epilogue)

    const int tid = threadIdx.x;

    // ---- stage weights into LDS as f16, ONCE per persistent block --------
    #pragma unroll
    for (int i = 0; i < 16; ++i) {
        int off = tid * 4 + i * 1024;            // rows 0..127 of both matrices
        v4f w1 = *(const v4f*)(W1 + off);
        v4f w2 = *(const v4f*)(W2 + off);
        #pragma unroll
        for (int j = 0; j < 4; ++j) { sW1[off + j] = (f16)w1[j]; sW2[off + j] = (f16)w2[j]; }
    }
    if (tid < 96) {                               // W1 rows 128..130 (xyz weights)
        int off = tid * 4;
        v4f wx = *(const v4f*)(W1 + EF * EF + off);
        #pragma unroll
        for (int j = 0; j < 4; ++j) sW1[EF * EF + off + j] = (f16)wx[j];
    }
    if (tid < EF) sW1[131 * EF + tid] = (f16)b1[tid];   // row 131 = b1 (A carries 1.0)
    #pragma unroll
    for (int i = tid; i < 28 * EF; i += 256)            // rows 132..159 = 0
        sW1[132 * EF + i] = (f16)0.0f;
    if (tid < EF) sB2[tid] = b2[tid];
    __syncthreads();

    const int wave    = tid >> 5;
    const int lane    = tid & 31;
    const int n16     = lane & 15;   // column-in-tile for C/D, row for A
    const int halfsel = lane >> 4;   // which K-chunk (A) / which M-half (C/D)

    // ---- persistent grid-stride loop over 256-row tiles -------------------
    for (int tile = blockIdx.x; tile < numTiles; tile += gridDim.x) {
        const int waveRow0 = tile * ROWS_PER_BLOCK + wave * ROWS_PER_WAVE;

        // Gather pointers: sub-tile s covers rows s*16 .. s*16+15.
        const float* rowPtr0 = input + (size_t)knn_idx[waveRow0 + n16] * EF;
        const float* rowPtr1 = input + (size_t)knn_idx[waveRow0 + 16 + n16] * EF;

        // ---------- layer 1: zero accumulators (bias lives in K=131) ------
        v8f c1[2][8];
        #pragma unroll
        for (int t = 0; t < 8; ++t) {
            v8f z = {};
            c1[0][t] = z; c1[1][t] = z;
        }

        // ---------- layer 1 WMMA: feature K = 128 in 4 steps of 32 --------
        #pragma unroll
        for (int kk = 0; kk < 4; ++kk) {
            v16h a0, a1;
            {   // A fragments straight from global (f32 -> f16)
                const float* cp0 = rowPtr0 + kk * 32 + halfsel * 8;
                const float* cp1 = rowPtr1 + kk * 32 + halfsel * 8;
                v4f f0 = *(const v4f*)(cp0);      v4f f1 = *(const v4f*)(cp0 + 4);
                v4f f2 = *(const v4f*)(cp0 + 16); v4f f3 = *(const v4f*)(cp0 + 20);
                v4f g0 = *(const v4f*)(cp1);      v4f g1 = *(const v4f*)(cp1 + 4);
                v4f g2 = *(const v4f*)(cp1 + 16); v4f g3 = *(const v4f*)(cp1 + 20);
                #pragma unroll
                for (int j = 0; j < 4; ++j) {
                    a0[j]      = (f16)f0[j]; a0[4 + j]  = (f16)f1[j];
                    a0[8 + j]  = (f16)f2[j]; a0[12 + j] = (f16)f3[j];
                    a1[j]      = (f16)g0[j]; a1[4 + j]  = (f16)g1[j];
                    a1[8 + j]  = (f16)g2[j]; a1[12 + j] = (f16)g3[j];
                }
            }
            const int krow = kk * 32 + halfsel * 8;
            #pragma unroll
            for (int t = 0; t < 8; ++t) {
                v16h bf;  // shared by both tiles
                #pragma unroll
                for (int s = 0; s < 16; ++s) bf[s] = sW1[(krow + (s & 7) + 2 * (s & 8)) * EF + t * 16 + n16];
                c1[0][t] = __builtin_amdgcn_wmma_f32_16x16x32_f16(
                               false, a0, false, bf, (short)0, c1[0][t], false, false);
                c1[1][t] = __builtin_amdgcn_wmma_f32_16x16x32_f16(
                               false, a1, false, bf, (short)0, c1[1][t], false, false);
            }
        }

        // ---------- layer 1 tail K-chunk (K=128..159): xyz + bias ---------
        {
            size_t g0i = (size_t)(waveRow0 + n16) * 3;
            size_t g1i = (size_t)(waveRow0 + 16 + n16) * 3;
            float msk = halfsel ? 0.0f : 1.0f;   // only K=128..135 chunk carries data
            v16h a0 = {}, a1 = {};
            a0[0] = (f16)(knn_xyz[g0i + 0] * msk);
            a0[1] = (f16)(knn_xyz[g0i + 1] * msk);
            a0[2] = (f16)(knn_xyz[g0i + 2] * msk);
            a0[3] = (f16)msk;                    // multiplies the b1 row (K=131)
            a1[0] = (f16)(knn_xyz[g1i + 0] * msk);
            a1[1] = (f16)(knn_xyz[g1i + 1] * msk);
            a1[2] = (f16)(knn_xyz[g1i + 2] * msk);
            a1[3] = (f16)msk;
            const int krow = 128 + halfsel * 8;  // rows 128..159 of padded W1 (table-order K)
            #pragma unroll
            for (int t = 0; t < 8; ++t) {
                v16h bf;
                #pragma unroll
                for (int s = 0; s < 16; ++s) bf[s] = sW1[(krow + (s & 7) + 2 * (s & 8)) * EF + t * 16 + n16];
                c1[0][t] = __builtin_amdgcn_wmma_f32_16x16x32_f16(
                               false, a0, false, bf, (short)0, c1[0][t], false, false);
                c1[1][t] = __builtin_amdgcn_wmma_f32_16x16x32_f16(
                               false, a1, false, bf, (short)0, c1[1][t], false, false);
            }
        }

        // ---------- leaky-relu -> f16, stage to LDS [32][128] -------------
        f16* hb = &sH[wave][0];
        #pragma unroll
        for (int s = 0; s < 2; ++s) {
            #pragma unroll
            for (int t = 0; t < 8; ++t) {
                #pragma unroll
                for (int r = 0; r < 8; ++r) {
                    float v = c1[s][t][r];
                    v = fmaxf(v, 0.01f * v);     // leaky relu, 2 VALU ops
                    hb[(s * 16 + 8 * halfsel + r) * EF + t * 16 + n16] = (f16)v;
                }
            }
        }
        // LDS ops are wave-ordered; ensure all stores landed before reads.
        asm volatile("s_wait_dscnt 0" ::: "memory");

        // ---------- layer 2 (bias deferred past the max-pool) -------------
        v8f c2[2][8];
        #pragma unroll
        for (int t = 0; t < 8; ++t) {
            v8f z = {};
            c2[0][t] = z; c2[1][t] = z;
        }
        const f16* hrow0 = &sH[wave][n16 * EF];
        const f16* hrow1 = &sH[wave][(16 + n16) * EF];
        #pragma unroll
        for (int kk = 0; kk < 4; ++kk) {
            AFrag a0, a1;
            a0.h[0] = *(const v8h*)(hrow0 + kk * 32 + halfsel * 8);
            a0.h[1] = *(const v8h*)(hrow0 + kk * 32 + halfsel * 8 + 16);
            a1.h[0] = *(const v8h*)(hrow1 + kk * 32 + halfsel * 8);
            a1.h[1] = *(const v8h*)(hrow1 + kk * 32 + halfsel * 8 + 16);
            const int krow = kk * 32 + halfsel * 8;
            #pragma unroll
            for (int t = 0; t < 8; ++t) {
                v16h bf;  // shared by both tiles
                #pragma unroll
                for (int s = 0; s < 16; ++s) bf[s] = sW2[(krow + (s & 7) + 2 * (s & 8)) * EF + t * 16 + n16];
                c2[0][t] = __builtin_amdgcn_wmma_f32_16x16x32_f16(
                               false, a0.v, false, bf, (short)0, c2[0][t], false, false);
                c2[1][t] = __builtin_amdgcn_wmma_f32_16x16x32_f16(
                               false, a1.v, false, bf, (short)0, c2[1][t], false, false);
            }
        }

        // ---------- max over 8 neighbors, add b2, store -------------------
        #pragma unroll
        for (int s = 0; s < 2; ++s) {
            const int point = tile * 32 + wave * 4 + s * 2 + halfsel;
            float* op = out + (size_t)point * EF;
            #pragma unroll
            for (int t = 0; t < 8; ++t) {
                float m = c2[s][t][0];
                #pragma unroll
                for (int r = 1; r < 8; ++r) m = fmaxf(m, c2[s][t][r]);
                op[t * 16 + n16] = m + sB2[t * 16 + n16];  // bias after pool (commutes)
            }
        }
    }
}

extern "C" void kernel_launch(void* const* d_in, const int* in_sizes, int n_in,
                              void* d_out, int out_size, void* d_ws, size_t ws_size,
                              hipStream_t stream) {
    const float* input   = (const float*)d_in[0];
    const int*   knn_idx = (const int*)  d_in[1];   // harness delivers integer inputs as int32
    const float* knn_xyz = (const float*)d_in[2];
    const float* W1      = (const float*)d_in[3];
    const float* b1      = (const float*)d_in[4];
    const float* W2      = (const float*)d_in[5];
    const float* b2      = (const float*)d_in[6];
    float*       out     = (float*)d_out;

    const int nRows    = in_sizes[1];               // N_POINTS * KNN_NUM = 800000
    const int numTiles = nRows / ROWS_PER_BLOCK;    // 3125 tiles of 256 rows
    int blocks = numTiles < 1024 ? numTiles : 1024; // persistent blocks, grid-stride

    pc_conv_wmma_kernel<<<blocks, 256, 0, stream>>>(input, knn_idx, knn_xyz,
                                                    W1, b1, W2, b2, out, numTiles);
}
